// ProgramExecutor_23029614641667
// MI455X (gfx1250) — matrix operand order repaired and checked by reading, not compiled, not run on hardware
//
#include <hip/hip_runtime.h>
#include <hip/hip_bf16.h>
#include <math.h>

#define HIDDEN   768
#define NPROG    64
#define BATCH    64
#define LENGTH   2048
#define PLEN     5
#define NTT      3
#define FINAL    (HIDDEN + NTT + 2)   // 773
#define ROWS     (PLEN * NPROG)       // 320
#define BROWPAD  776                  // 768 + 8 bf16 pad -> conflict-free ds reads
#define MTILE    256                  // L-rows per workgroup (32 per wave)

typedef __attribute__((ext_vector_type(16))) __bf16 v16bf;
typedef __attribute__((ext_vector_type(8)))  float  v8f;

// ---------- monotone float <-> uint mapping for atomicMax on f32 ----------
__device__ __forceinline__ unsigned int fmap(float f) {
    unsigned int u = __float_as_uint(f);
    return (u & 0x80000000u) ? ~u : (u | 0x80000000u);
}
__device__ __forceinline__ float funmap(unsigned int u) {
    unsigned int v = (u & 0x80000000u) ? (u & 0x7fffffffu) : ~u;
    return __uint_as_float(v);
}

// ---------------- workspace layout ----------------
#define WS_VEC_BYTES  (ROWS * HIDDEN * 2)
#define WS_TTLOG_OFF  (WS_VEC_BYTES)
#define WS_BIAS_OFF   (WS_TTLOG_OFF + NPROG * NTT * 4)
#define WS_SCALE_OFF  (WS_BIAS_OFF + ROWS * 4)
#define WS_SELMAX_OFF (WS_SCALE_OFF + ROWS * 4)
#define SELMAX_N      (PLEN * BATCH * NPROG)

// =====================================================================
// Kernel 0: zero the selmax scratch
// =====================================================================
__global__ void k_init(unsigned int* selmax) {
    int i = blockIdx.x * blockDim.x + threadIdx.x;
    if (i < SELMAX_N) selmax[i] = 0u;   // maps to "smaller than any real"
}

// =====================================================================
// Kernel 1: p = program @ W.T + b  (320 x 773, K=768)
// =====================================================================
__global__ void k_setup(const float* __restrict__ program,
                        const float* __restrict__ W,
                        const float* __restrict__ bvec,
                        __bf16* __restrict__ vecb,
                        float* __restrict__ ttlog,
                        float* __restrict__ biasA,
                        float* __restrict__ scaleA) {
    __shared__ float prog_s[8][HIDDEN];
    const int r0   = blockIdx.x * 8;
    const int tid  = threadIdx.x;
    for (int idx = tid; idx < 8 * HIDDEN; idx += 256) {
        int r = idx / HIDDEN, k = idx % HIDDEN;
        prog_s[r][k] = program[(size_t)(r0 + r) * HIDDEN + k];
    }
    __syncthreads();

    const int wave = tid >> 5;
    const int lane = tid & 31;

    for (int f = wave; f < FINAL; f += 8) {
        float acc[8] = {0.f, 0.f, 0.f, 0.f, 0.f, 0.f, 0.f, 0.f};
        const float* wrow = W + (size_t)f * HIDDEN;
        for (int k = lane; k < HIDDEN; k += 32) {
            float wv = wrow[k];
#pragma unroll
            for (int r = 0; r < 8; ++r) acc[r] += wv * prog_s[r][k];
        }
#pragma unroll
        for (int r = 0; r < 8; ++r) {
            float a = acc[r];
#pragma unroll
            for (int off = 16; off >= 1; off >>= 1) a += __shfl_xor(a, off, 32);
            acc[r] = a;
        }
        if (lane == 0) {
            const float bb = bvec[f];
#pragma unroll
            for (int r = 0; r < 8; ++r) {
                const int row = r0 + r;          // row = s*64 + n
                const int s = row >> 6, n = row & 63;
                const float val = acc[r] + bb;
                if (f < HIDDEN) {
                    vecb[(size_t)row * HIDDEN + f] = (__bf16)val;
                } else if (f < HIDDEN + NTT) {
                    if (s == 0) ttlog[n * NTT + (f - HIDDEN)] = val;
                } else if (f == HIDDEN + NTT) {
                    biasA[row] = val;
                } else {
                    scaleA[row] = val;
                }
            }
        }
    }
}

// =====================================================================
// Kernel 2: sel[s,b,l,p] = (inputs[b,l,:] . vec[s,p,:]) / sigma
//   WMMA bf16; B staged in LDS via async global->LDS copies;
//   each wave computes 32 L-rows so every B fragment feeds 2 WMMAs.
// =====================================================================
__device__ __forceinline__ void cvt4(v16bf& v, int base, float4 f) {
    v[base + 0] = (__bf16)f.x;
    v[base + 1] = (__bf16)f.y;
    v[base + 2] = (__bf16)f.z;
    v[base + 3] = (__bf16)f.w;
}

__device__ __forceinline__ v16bf load_a_frag(const float* arow, int kb) {
    float4 a0 = *(const float4*)(arow + kb);
    float4 a1 = *(const float4*)(arow + kb + 4);
    float4 a2 = *(const float4*)(arow + kb + 16);
    float4 a3 = *(const float4*)(arow + kb + 20);
    v16bf af;
    cvt4(af, 0,  a0);
    cvt4(af, 4,  a1);
    cvt4(af, 8,  a2);
    cvt4(af, 12, a3);
    return af;
}

extern __shared__ __bf16 bs[];   // NPROG * BROWPAD bf16 = 99328 B

__global__ void __launch_bounds__(256)
k_selgemm(const float* __restrict__ inputs,
          const __bf16* __restrict__ vecb,
          float* __restrict__ out_sel,     // d_out + 4096, layout (s,b,l,p)
          unsigned int* __restrict__ selmax) {
    const int ltiles = LENGTH / MTILE;          // 8
    const int b    = blockIdx.x / ltiles;
    const int l0   = (blockIdx.x % ltiles) * MTILE;
    const int tid  = threadIdx.x;
    const int wv   = tid >> 5;                  // 0..7
    const int lane = tid & 31;
    const int half = lane >> 4;                 // 0|1
    const int lr   = lane & 15;

    const float invsigma = 1.0f / (sqrtf((float)HIDDEN) * 6.0f);

    // A-matrix rows for this lane (lanes map M = lr per ISA layout)
    const float* arow0 =
        inputs + ((size_t)b * LENGTH + (size_t)(l0 + wv * 32 + lr)) * HIDDEN;
    const float* arow1 = arow0 + (size_t)16 * HIDDEN;

    for (int s = 0; s < PLEN; ++s) {
        // ---- stage vec[s] (64x768 bf16) into padded LDS, async path ----
        __syncthreads();   // previous iteration's readers are done
        const __bf16* gsrc = vecb + (size_t)s * NPROG * HIDDEN;
        for (int c = tid; c < NPROG * (HIDDEN / 8); c += 256) {
            const int n  = c / (HIDDEN / 8);
            const int kc = (c % (HIDDEN / 8)) * 8;
            const unsigned int lds_byte =
                (unsigned int)((n * BROWPAD + kc) * 2);
            const unsigned long long gaddr =
                (unsigned long long)(size_t)(gsrc + (size_t)n * HIDDEN + kc);
            asm volatile("global_load_async_to_lds_b128 %0, %1, off"
                         :: "v"(lds_byte), "v"(gaddr)
                         : "memory");
        }
        asm volatile("s_wait_asynccnt 0x0" ::: "memory");
        __syncthreads();

        v8f acc[8] = {};   // [mfrag*4 + nt]

        for (int kt = 0; kt < HIDDEN / 32; ++kt) {
            const int k0 = kt * 32;
            const int kb = k0 + half * 8;

            __builtin_prefetch(arow0 + k0 + 64, 0, 0);
            __builtin_prefetch(arow1 + k0 + 64, 0, 0);

            // --- two A fragments (rows +0 and +16), f32 -> bf16 ---
            v16bf af0 = load_a_frag(arow0, kb);
            v16bf af1 = load_a_frag(arow1, kb);

#pragma unroll
            for (int nt = 0; nt < 4; ++nt) {
                // --- B fragment from LDS (conflict-free padded rows) ---
                const __bf16* bp =
                    &bs[(nt * 16 + lr) * BROWPAD + k0 + half * 8];
                union { float4 q[2]; v16bf v; } bu;
                bu.q[0] = *(const float4*)(bp);
                bu.q[1] = *(const float4*)(bp + 16);

                acc[nt] = __builtin_amdgcn_wmma_f32_16x16x32_bf16(
                    false, af0, false, bu.v, (short)0, acc[nt],
                    false, false);
                acc[4 + nt] = __builtin_amdgcn_wmma_f32_16x16x32_bf16(
                    false, af1, false, bu.v, (short)0, acc[4 + nt],
                    false, false);
            }
        }

        // --- scale, store, running max over this wave's 32 rows ---
#pragma unroll
        for (int nt = 0; nt < 4; ++nt) {
            float mx = -3.4e38f;
#pragma unroll
            for (int mf = 0; mf < 2; ++mf) {
                const size_t selrow =
                    ((size_t)(s * BATCH + b) * LENGTH +
                     (size_t)(l0 + wv * 32 + mf * 16)) * NPROG;
#pragma unroll
                for (int r = 0; r < 8; ++r) {
                    const float v = acc[mf * 4 + nt][r] * invsigma;
                    const int m = half * 8 + r;   // row within 16-row frag
                    out_sel[selrow + (size_t)m * NPROG + nt * 16 + lr] = v;
                    mx = fmaxf(mx, v);
                }
            }
            const float o = __shfl_xor(mx, 16, 32);
            mx = fmaxf(mx, o);
            if (half == 0) {
                atomicMax(&selmax[(s * BATCH + b) * NPROG + nt * 16 + lr],
                          fmap(mx));
            }
        }
    }
}

// =====================================================================
// Kernel 3: gate recurrence + scores.  One thread per (batch, program).
// =====================================================================
__global__ void k_final(const unsigned int* __restrict__ selmax,
                        const float* __restrict__ ttlog,
                        const float* __restrict__ biasA,
                        const float* __restrict__ scaleA,
                        float* __restrict__ scores) {
    const int idx = blockIdx.x * blockDim.x + threadIdx.x;
    if (idx >= BATCH * NPROG) return;
    const int b = idx >> 6;
    const int p = idx & 63;

    const float sigma = sqrtf((float)HIDDEN) * 6.0f;

    // template index table t[tt][s]
    const int T[NTT][PLEN] = {
        {0, 4, 4, 4, 4},
        {0, 0, 1, 4, 4},
        {0, 0, 1, 0, 1}
    };

    float sel[PLEN];
#pragma unroll
    for (int s = 0; s < PLEN; ++s)
        sel[s] = funmap(selmax[(s * BATCH + b) * NPROG + p]);

    float hist[PLEN][NTT];
    for (int s = 0; s < PLEN; ++s) {
        float a1h[NTT], a2h[NTT];
        if (s == 0) {
            for (int t = 0; t < NTT; ++t) { a1h[t] = 0.f; a2h[t] = 0.f; }
        } else {
            for (int t = 0; t < NTT; ++t) {
                float lg[PLEN], m = -3.4e38f;
                for (int i = 0; i < s; ++i) {
                    float q = 0.f;
                    if (s == 2 && t >= 1 && i == 0) q = 1000.f;
                    if (s == 4 && t == 2 && i == 2) q = 1000.f;
                    lg[i] = q / sigma;
                    m = fmaxf(m, lg[i]);
                }
                float sum = 0.f, w[PLEN];
                for (int i = 0; i < s; ++i) { w[i] = expf(lg[i] - m); sum += w[i]; }
                float a1 = 0.f;
                for (int i = 0; i < s; ++i) a1 += (w[i] / sum) * hist[i][t];
                a1h[t] = a1;
                a2h[t] = hist[s - 1][t];
            }
        }
        for (int t = 0; t < NTT; ++t) {
            float v;
            switch (T[t][s]) {
                case 0:  v = sel[s];                      break;
                case 1:  v = fminf(a1h[t], a2h[t]);       break;
                case 2:  v = fmaxf(a1h[t], a2h[t]);       break;
                case 3:  v = -a2h[t];                     break;
                case 4:  v = a2h[t];                      break;
                default: v = fmaxf(-a1h[t], a2h[t]);      break;
            }
            hist[s][t] = v;
        }
    }

    // type_index = softmax over NTT of ttlog[p]
    float t0 = ttlog[p * NTT + 0], t1 = ttlog[p * NTT + 1], t2 = ttlog[p * NTT + 2];
    float m = fmaxf(t0, fmaxf(t1, t2));
    float e0 = expf(t0 - m), e1 = expf(t1 - m), e2 = expf(t2 - m);
    float inv = 1.0f / (e0 + e1 + e2);
    float ti[NTT] = { e0 * inv, e1 * inv, e2 * inv };

    const float sc = scaleA[p];   // scale[0][p]
    const float bi = biasA[p];    // bias[0][p]
    float score = 0.f;
#pragma unroll
    for (int t = 0; t < NTT; ++t)
        score += (hist[PLEN - 1][t] * sc + bi) * ti[t];

    scores[b * NPROG + p] = score;
}

// =====================================================================
extern "C" void kernel_launch(void* const* d_in, const int* in_sizes, int n_in,
                              void* d_out, int out_size, void* d_ws, size_t ws_size,
                              hipStream_t stream) {
    const float* inputs  = (const float*)d_in[0];   // (64,2048,768)
    const float* program = (const float*)d_in[1];   // (5,64,768)
    const float* W       = (const float*)d_in[2];   // (773,768)
    const float* bvec    = (const float*)d_in[3];   // (773,)

    float* out = (float*)d_out;
    float* out_scores = out;                        // (64,64)
    float* out_sel    = out + BATCH * NPROG;        // (5,64,2048,64)

    char* ws = (char*)d_ws;
    __bf16*       vecb   = (__bf16*)ws;
    float*        ttlog  = (float*)(ws + WS_TTLOG_OFF);
    float*        biasA  = (float*)(ws + WS_BIAS_OFF);
    float*        scaleA = (float*)(ws + WS_SCALE_OFF);
    unsigned int* selmax = (unsigned int*)(ws + WS_SELMAX_OFF);

    k_init<<<(SELMAX_N + 255) / 256, 256, 0, stream>>>(selmax);

    k_setup<<<ROWS / 8, 256, 0, stream>>>(program, W, bvec,
                                          vecb, ttlog, biasA, scaleA);

    const size_t lds_bytes = (size_t)NPROG * BROWPAD * 2;   // 99328
    k_selgemm<<<BATCH * (LENGTH / MTILE), 256, lds_bytes, stream>>>(
        inputs, vecb, out_sel, selmax);

    k_final<<<(BATCH * NPROG + 255) / 256, 256, 0, stream>>>(selmax, ttlog,
                                                             biasA, scaleA,
                                                             out_scores);
}
